// SGFormerAttention_8959301780006
// MI455X (gfx1250) — matrix operand order repaired and checked by reading, not compiled, not run on hardware
//
#include <hip/hip_runtime.h>
#include <hip/hip_bf16.h>

// ---------------------------------------------------------------------------
// SGFormer linear attention for MI455X (gfx1250, wave32, WMMA bf16 + TDM)
// ---------------------------------------------------------------------------

#define B_  2
#define N_  32768
#define C_  1024
#define H_  8
#define D_  128
#define EPS_ 1e-6f

typedef __attribute__((ext_vector_type(8)))  float        v8f;
typedef __attribute__((ext_vector_type(16))) __bf16       v16bf;
typedef __attribute__((ext_vector_type(8)))  __bf16       v8bf;
typedef __attribute__((ext_vector_type(4)))  unsigned int u32x4;
typedef __attribute__((ext_vector_type(8)))  int          i32x8;

static __device__ inline v16bf cat16(v8bf a, v8bf b) {
  return __builtin_shufflevector(a, b, 0, 1, 2, 3, 4, 5, 6, 7,
                                        8, 9, 10, 11, 12, 13, 14, 15);
}

static __device__ inline v8f wmma_bf16(v16bf a, v16bf b, v8f c) {
  // D(16x16,f32) = A(16x32,bf16) * B(32x16,bf16) + C
  return __builtin_amdgcn_wmma_f32_16x16x32_bf16(
      /*neg_a=*/false, a, /*neg_b=*/false, b,
      /*c_mod=*/(short)0, c, /*reuse_a=*/false, /*reuse_b=*/false);
}

// ---------------------------------------------------------------------------
// Tensor Data Mover: 2D tile load Global -> LDS (bf16 elements).
// Builds a D# per CDNA5 ISA ch.8 and issues the <=2D form (VADDR2/3 = NULL).
// Caller must be a single wave; follow with s_wait_tensorcnt(0) + barrier.
// ---------------------------------------------------------------------------
static __device__ inline void tdm_load_2d_bf16(unsigned lds_off,
                                               const void* gptr,
                                               unsigned tile_w,      // elems
                                               unsigned tile_h,      // rows
                                               unsigned stride_elems)
{
  unsigned long long ga = (unsigned long long)gptr;
  u32x4 g0;
  g0[0] = 1u;                                   // count=1, no gather/restore
  g0[1] = lds_off;                              // lds_addr (bytes)
  g0[2] = (unsigned)(ga & 0xFFFFFFFFull);       // global_addr[31:0]
  g0[3] = (unsigned)((ga >> 32) & 0x1FFFFFFull) // global_addr[56:32]
          | (2u << 30);                         // type=2 ("image")
  // group1: data_size=1 (2 bytes); tensor dims huge (tile known in-bounds);
  // tile dims + dim0 stride; no padding/iterate/multicast/atomic-barrier.
  const unsigned TD = 0x40000000u;              // large tensor_dim0/1
  i32x8 g1;
  g1[0] = (int)(1u << 16);                      // wg_mask=0 | data_size=1
  g1[1] = (int)((TD & 0xFFFFu) << 16);          // [63:48]=tensor_dim0[15:0]
  g1[2] = (int)(((TD >> 16) & 0xFFFFu)          // [79:64]=tensor_dim0[31:16]
          | ((TD & 0xFFFFu) << 16));            // [95:80]=tensor_dim1[15:0]
  g1[3] = (int)(((TD >> 16) & 0xFFFFu)          // [111:96]=tensor_dim1[31:16]
          | (tile_w << 16));                    // [127:112]=tile_dim0
  g1[4] = (int)tile_h;                          // [143:128]=tile_dim1, dim2=0
  g1[5] = (int)stride_elems;                    // tensor_dim0_stride[31:0]
  g1[6] = 0;                                    // stride0[47:32], stride1 lo
  g1[7] = 0;
  asm volatile("tensor_load_to_lds %0, %1" :: "s"(g0), "s"(g1) : "memory");
}

// ---------------------------------------------------------------------------
// K0a: f32 -> bf16 convert, 8 elements per iteration
// ---------------------------------------------------------------------------
__global__ void k_cvt8(const float* __restrict__ src, __bf16* __restrict__ dst,
                       size_t n8) {
  size_t i = (size_t)blockIdx.x * blockDim.x + threadIdx.x;
  size_t stride = (size_t)gridDim.x * blockDim.x;
  for (; i < n8; i += stride) {
    const float4* s = (const float4*)(src + i * 8);
    float4 f0 = s[0], f1 = s[1];
    v8bf o;
    o[0] = (__bf16)f0.x; o[1] = (__bf16)f0.y;
    o[2] = (__bf16)f0.z; o[3] = (__bf16)f0.w;
    o[4] = (__bf16)f1.x; o[5] = (__bf16)f1.y;
    o[6] = (__bf16)f1.z; o[7] = (__bf16)f1.w;
    *(v8bf*)(dst + i * 8) = o;
  }
}

// K0b: zero f32 region (kv + ksum accumulators)
__global__ void k_zero(float* __restrict__ p, size_t n) {
  size_t i = (size_t)blockIdx.x * blockDim.x + threadIdx.x;
  size_t stride = (size_t)gridDim.x * blockDim.x;
  for (; i < n; i += stride) p[i] = 0.0f;
}

// ---------------------------------------------------------------------------
// K1: fused QKV GEMM (bf16 WMMA) + eps fixup + L2 normalize (q,k) + mask (v)
//   grid = (B * N/64, H), block = 256 (8 waves)
//   x tile staged via Tensor Data Mover (wave 0); W tiles transposed manually.
// ---------------------------------------------------------------------------
__global__ __launch_bounds__(256) void k_qkv(
    const __bf16* __restrict__ xbf,   // [B][N][C] bf16
    const __bf16* __restrict__ wbf,   // [3][C][C] bf16 (q,k,v packed)
    const unsigned char* __restrict__ mask,  // [B][N] bool
    __bf16* __restrict__ qbf,         // [B*H][N][D] bf16 (normalized)
    __bf16* __restrict__ kbf,         // [B*H][N][D] bf16 (normalized)
    float*  __restrict__ vf)          // [B*H][N][D] f32  (masked)
{
  __shared__ union {
    struct {
      __bf16 xs[64][32];       // x tile, row-major: [row][k]  (TDM dest)
      __bf16 wT[3 * 128][32];  // W tile transposed: [proj*128 + col][k]
    } in;
    float stg[64][128];        // post-GEMM staging (aliases GEMM tiles)
  } u;
  __shared__ float red[64][4];
  __shared__ float nrm[64];

  const int tid  = threadIdx.x;
  const int lane = tid & 31;
  const int w    = tid >> 5;                 // wave id 0..7 -> 16-col strip
  const int b    = blockIdx.x >> 9;          // 512 row-tiles per batch
  const int tile = blockIdx.x & 511;
  const int h    = blockIdx.y;
  const int n0   = tile * 64;

  const int lm  = lane & 15;
  const int kbA = (lane < 16) ? 0 : 8;       // A operand K base (per ISA layout)
  const int kbB = (lane < 16) ? 0 : 8;       // B operand K base

  const unsigned xs_off = (unsigned)(unsigned long long)(void*)&u.in.xs[0][0];

  v8f acc[3][4] = {};                         // [proj][row-block], 96 VGPRs

  for (int k0 = 0; k0 < C_; k0 += 32) {
    __syncthreads();
    // ---- stage x tile [64 rows][32 k] via TDM (wave 0 only) ----
    if (w == 0) {
      tdm_load_2d_bf16(xs_off, xbf + ((size_t)b * N_ + n0) * C_ + k0,
                       /*tile_w=*/32, /*tile_h=*/64, /*stride=*/C_);
    }
    // ---- stage W tiles transposed: wT[p*128 + col][k] (all waves) ----
    #pragma unroll
    for (int p = 0; p < 3; ++p) {
      int kk = tid >> 3, cgrp = (tid & 7) * 16;
      const __bf16* g = wbf + ((size_t)p << 20) +
                        (size_t)(k0 + kk) * C_ + h * D_ + cgrp;
      v8bf a0 = *(const v8bf*)g;
      v8bf a1 = *(const v8bf*)(g + 8);
      #pragma unroll
      for (int i = 0; i < 8; ++i) {
        u.in.wT[p * 128 + cgrp + i][kk]     = a0[i];
        u.in.wT[p * 128 + cgrp + 8 + i][kk] = a1[i];
      }
    }
    if (w == 0) __builtin_amdgcn_s_wait_tensorcnt(0);
    __syncthreads();
    // ---- WMMA: 4 row-blocks x 3 projections per wave ----
    v16bf A[4];
    #pragma unroll
    for (int rb = 0; rb < 4; ++rb) {
      v8bf lo = *(const v8bf*)&u.in.xs[16 * rb + lm][kbA];
      v8bf hi = *(const v8bf*)&u.in.xs[16 * rb + lm][kbA + 16];
      A[rb] = cat16(lo, hi);
    }
    #pragma unroll
    for (int p = 0; p < 3; ++p) {
      v8bf blo = *(const v8bf*)&u.in.wT[p * 128 + 16 * w + lm][kbB];
      v8bf bhi = *(const v8bf*)&u.in.wT[p * 128 + 16 * w + lm][kbB + 16];
      v16bf Bv = cat16(blo, bhi);
      #pragma unroll
      for (int rb = 0; rb < 4; ++rb)
        acc[p][rb] = wmma_bf16(A[rb], Bv, acc[p][rb]);
    }
  }

  // ---- post-process each projection via LDS staging ----
  #pragma unroll
  for (int p = 0; p < 3; ++p) {
    const bool isV = (p == 2);
    __syncthreads();  // prior readers of staging region done
    #pragma unroll
    for (int rb = 0; rb < 4; ++rb)
      #pragma unroll
      for (int i = 0; i < 8; ++i) {
        int row = 16 * rb + i + ((lane >= 16) ? 8 : 0);
        float val = acc[p][rb][i];
        if (!isV && val == 0.0f) val = EPS_;   // eps fixup before norm
        u.stg[row][16 * w + lm] = val;
      }
    __syncthreads();
    if (!isV) {
      // L2 norm over 128 channels per row
      {
        int row = tid >> 2, q4 = tid & 3;
        float s = 0.0f;
        for (int j = 0; j < 32; ++j) {
          float v = u.stg[row][q4 * 32 + j];
          s += v * v;
        }
        red[row][q4] = s;
      }
      __syncthreads();
      if (tid < 64)
        nrm[tid] = rsqrtf(red[tid][0] + red[tid][1] + red[tid][2] + red[tid][3]);
      __syncthreads();
      __bf16* dst = (p == 0 ? qbf : kbf) + ((size_t)(b * H_ + h) * N_ + n0) * D_;
      #pragma unroll
      for (int j = 0; j < 4; ++j) {
        int c = j * 256 + tid, row = c >> 4, cg = (c & 15) * 8;
        float sc = nrm[row];
        v8bf o;
        #pragma unroll
        for (int e = 0; e < 8; ++e) o[e] = (__bf16)(u.stg[row][cg + e] * sc);
        *(v8bf*)(dst + (size_t)row * D_ + cg) = o;
      }
    } else {
      float* dst = vf + ((size_t)(b * H_ + h) * N_ + n0) * D_;
      #pragma unroll
      for (int j = 0; j < 8; ++j) {
        int c = j * 256 + tid, row = c >> 5, cg = (c & 31) * 4;
        float m = mask[(size_t)b * N_ + n0 + row] ? 1.0f : 0.0f;
        float4 o;
        o.x = u.stg[row][cg + 0] * m;
        o.y = u.stg[row][cg + 1] * m;
        o.z = u.stg[row][cg + 2] * m;
        o.w = u.stg[row][cg + 3] * m;
        *(float4*)(dst + (size_t)row * D_ + cg) = o;
      }
    }
  }
}

// ---------------------------------------------------------------------------
// K2: kv[b,h] += k^T v over an n-chunk (WMMA, A = k^T via LDS transpose);
//     ks_sum folded in. grid = (B*H, 64 chunks of 512 n), block = 256.
// ---------------------------------------------------------------------------
__global__ __launch_bounds__(256) void k_kv(
    const __bf16* __restrict__ kbf, const float* __restrict__ vf,
    float* __restrict__ kv,          // [B*H][D][D] f32 (atomically accumulated)
    float* __restrict__ ksum)        // [B*H][D]    f32
{
  __shared__ __bf16 kT[128][32];     // kT[m][n_local]
  __shared__ __bf16 vT[128][32];     // vT[d][n_local]

  const int tid  = threadIdx.x;
  const int lane = tid & 31;
  const int w    = tid >> 5;         // wave -> 16-row M strip
  const int bh   = blockIdx.x;
  const int n0   = blockIdx.y * 512;
  const int lm   = lane & 15;
  const int kbA  = (lane < 16) ? 0 : 8;
  const int kbB  = (lane < 16) ? 0 : 8;

  v8f acc[8] = {};
  float ksumLoc = 0.0f;

  for (int s = 0; s < 16; ++s) {     // 512 / 32
    const int nb0 = n0 + s * 32;
    __syncthreads();
    {  // stage k transposed
      int nl = tid >> 3, mg = (tid & 7) * 16;
      const __bf16* g = kbf + ((size_t)bh * N_ + nb0 + nl) * D_ + mg;
      v8bf a0 = *(const v8bf*)g, a1 = *(const v8bf*)(g + 8);
      #pragma unroll
      for (int i = 0; i < 8; ++i) {
        kT[mg + i][nl]     = a0[i];
        kT[mg + 8 + i][nl] = a1[i];
      }
    }
    {  // stage v transposed (f32 -> bf16 on the fly)
      int nl = tid >> 3, dg = (tid & 7) * 16;
      const float4* g =
          (const float4*)(vf + ((size_t)bh * N_ + nb0 + nl) * D_ + dg);
      #pragma unroll
      for (int c4 = 0; c4 < 4; ++c4) {
        float4 f = g[c4];
        vT[dg + c4 * 4 + 0][nl] = (__bf16)f.x;
        vT[dg + c4 * 4 + 1][nl] = (__bf16)f.y;
        vT[dg + c4 * 4 + 2][nl] = (__bf16)f.z;
        vT[dg + c4 * 4 + 3][nl] = (__bf16)f.w;
      }
    }
    __syncthreads();
    if (tid < 128) {  // ks_sum partial from staged tile
      #pragma unroll
      for (int nl = 0; nl < 32; ++nl) ksumLoc += (float)kT[tid][nl];
    }
    v8bf lo = *(const v8bf*)&kT[16 * w + lm][kbA];
    v8bf hi = *(const v8bf*)&kT[16 * w + lm][kbA + 16];
    v16bf A = cat16(lo, hi);
    #pragma unroll
    for (int nb = 0; nb < 8; ++nb) {
      v8bf blo = *(const v8bf*)&vT[16 * nb + lm][kbB];
      v8bf bhi = *(const v8bf*)&vT[16 * nb + lm][kbB + 16];
      acc[nb] = wmma_bf16(A, cat16(blo, bhi), acc[nb]);
    }
  }

  float* kvp = kv + (size_t)bh * D_ * D_;
  #pragma unroll
  for (int nb = 0; nb < 8; ++nb)
    #pragma unroll
    for (int i = 0; i < 8; ++i) {
      int m = 16 * w + i + ((lane >= 16) ? 8 : 0);
      int d = 16 * nb + lm;
      atomicAdd(&kvp[m * D_ + d], acc[nb][i]);
    }
  if (tid < 128) atomicAdd(&ksum[(size_t)bh * D_ + tid], ksumLoc);
}

// ---------------------------------------------------------------------------
// K3: out = mean_h ( (q @ kv + n*v) / (q . ksum + n) )
//     grid = B * N/64, block = 256; loops over heads, accumulates in regs.
//     q tile staged via TDM.
// ---------------------------------------------------------------------------
__global__ __launch_bounds__(256) void k_out(
    const __bf16* __restrict__ qbf, const float* __restrict__ vf,
    const float* __restrict__ kv, const float* __restrict__ ksum,
    float* __restrict__ out)
{
  __shared__ __bf16 qT[64][128];                       // q tile (TDM dest)
  __shared__ union {
    __bf16 kvT[128][128];                              // kv transposed (B op)
    float  sArr[64][128];                              // q@kv result staging
  } u;
  __shared__ float red[64][4];
  __shared__ float zrow[64];
  __shared__ float ksumS[128];

  const int tid  = threadIdx.x;
  const int lane = tid & 31;
  const int w    = tid >> 5;
  const int b    = blockIdx.x >> 9;
  const int tile = blockIdx.x & 511;
  const int n0   = tile * 64;
  const int lm   = lane & 15;
  const int kbA  = (lane < 16) ? 0 : 8;
  const int kbB  = (lane < 16) ? 0 : 8;
  const float nF = (float)N_;

  const unsigned qT_off = (unsigned)(unsigned long long)(void*)&qT[0][0];

  float accOut[32];
  #pragma unroll
  for (int j = 0; j < 32; ++j) accOut[j] = 0.0f;

  for (int h = 0; h < H_; ++h) {
    const int bh = b * H_ + h;
    __syncthreads();  // previous head's readers done
    // stage q tile [64][128] via TDM (wave 0)
    if (w == 0) {
      tdm_load_2d_bf16(qT_off, qbf + ((size_t)bh * N_ + n0) * D_,
                       /*tile_w=*/128, /*tile_h=*/64, /*stride=*/D_);
    }
    {  // stage kv transposed, f32 -> bf16 (all waves)
      int m = tid >> 1, dg = (tid & 1) * 64;
      const float4* g = (const float4*)(kv + ((size_t)bh * D_ + m) * D_ + dg);
      #pragma unroll
      for (int c4 = 0; c4 < 16; ++c4) {
        float4 f = g[c4];
        u.kvT[dg + c4 * 4 + 0][m] = (__bf16)f.x;
        u.kvT[dg + c4 * 4 + 1][m] = (__bf16)f.y;
        u.kvT[dg + c4 * 4 + 2][m] = (__bf16)f.z;
        u.kvT[dg + c4 * 4 + 3][m] = (__bf16)f.w;
      }
    }
    if (tid < 128) ksumS[tid] = ksum[(size_t)bh * D_ + tid];
    if (w == 0) __builtin_amdgcn_s_wait_tensorcnt(0);
    __syncthreads();
    {  // normalizer: z[row] = q[row,:] . ksum + n
      int row = tid >> 2, q4 = tid & 3;
      float sdot = 0.0f;
      for (int j = 0; j < 32; ++j)
        sdot += (float)qT[row][q4 * 32 + j] * ksumS[q4 * 32 + j];
      red[row][q4] = sdot;
    }
    __syncthreads();
    if (tid < 64)
      zrow[tid] = red[tid][0] + red[tid][1] + red[tid][2] + red[tid][3] + nF;

    // GEMM: s(64x128) = qT(64x128) @ kv(128x128)
    v8f acc[4] = {};
    #pragma unroll
    for (int k0 = 0; k0 < 128; k0 += 32) {
      v16bf A[4];
      #pragma unroll
      for (int rb = 0; rb < 4; ++rb) {
        v8bf lo = *(const v8bf*)&qT[16 * rb + lm][k0 + kbA];
        v8bf hi = *(const v8bf*)&qT[16 * rb + lm][k0 + kbA + 16];
        A[rb] = cat16(lo, hi);
      }
      v8bf blo = *(const v8bf*)&u.kvT[16 * w + lm][k0 + kbB];
      v8bf bhi = *(const v8bf*)&u.kvT[16 * w + lm][k0 + kbB + 16];
      v16bf Bv = cat16(blo, bhi);
      #pragma unroll
      for (int rb = 0; rb < 4; ++rb) acc[rb] = wmma_bf16(A[rb], Bv, acc[rb]);
    }
    __syncthreads();  // kvT reads done before aliasing sArr over it
    #pragma unroll
    for (int rb = 0; rb < 4; ++rb)
      #pragma unroll
      for (int i = 0; i < 8; ++i) {
        int row = 16 * rb + i + ((lane >= 16) ? 8 : 0);
        u.sArr[row][16 * w + lm] = acc[rb][i];
      }
    __syncthreads();
    #pragma unroll
    for (int j = 0; j < 32; ++j) {
      int idx = j * 256 + tid, row = idx >> 7, col = idx & 127;
      float vv = vf[((size_t)bh * N_ + n0 + row) * D_ + col];
      accOut[j] += (u.sArr[row][col] + nF * vv) / zrow[row];
    }
  }

  #pragma unroll
  for (int j = 0; j < 32; ++j) {
    int idx = j * 256 + tid, row = idx >> 7, col = idx & 127;
    out[((size_t)b * N_ + n0 + row) * D_ + col] = accOut[j] * (1.0f / H_);
  }
}

// ---------------------------------------------------------------------------
// Launch
// ---------------------------------------------------------------------------
extern "C" void kernel_launch(void* const* d_in, const int* in_sizes, int n_in,
                              void* d_out, int out_size, void* d_ws,
                              size_t ws_size, hipStream_t stream) {
  (void)in_sizes; (void)n_in; (void)out_size; (void)ws_size;

  const float* x  = (const float*)d_in[0];
  const unsigned char* mask = (const unsigned char*)d_in[1];
  const float* Wq = (const float*)d_in[2];
  const float* Wk = (const float*)d_in[3];
  const float* Wv = (const float*)d_in[4];
  float* out = (float*)d_out;

  // workspace layout (bytes)
  const size_t X_ELEMS  = (size_t)B_ * N_ * C_;        // 67,108,864
  const size_t W_ELEMS  = (size_t)C_ * C_;             // 1,048,576 per proj
  const size_t P_ELEMS  = (size_t)B_ * H_ * N_ * D_;   // 67,108,864
  char* ws = (char*)d_ws;
  __bf16* xbf  = (__bf16*)(ws);                                   // 128 MB
  __bf16* wbf  = (__bf16*)(ws + X_ELEMS * 2);                     //   6 MB
  __bf16* qbf  = (__bf16*)(ws + X_ELEMS * 2 + 3 * W_ELEMS * 2);   // 128 MB
  __bf16* kbf  = (__bf16*)((char*)qbf + P_ELEMS * 2);             // 128 MB
  float*  vf   = (float*) ((char*)kbf + P_ELEMS * 2);             // 256 MB
  float*  kv   = (float*) ((char*)vf  + P_ELEMS * 4);             //   1 MB
  float*  ksum = (float*) ((char*)kv  + (size_t)B_ * H_ * D_ * D_ * 4);

  // K0: converts + zero accumulators
  k_cvt8<<<4096, 256, 0, stream>>>(x,  xbf, X_ELEMS / 8);
  k_cvt8<<<512, 256, 0, stream>>>(Wq, wbf + 0 * W_ELEMS, W_ELEMS / 8);
  k_cvt8<<<512, 256, 0, stream>>>(Wk, wbf + 1 * W_ELEMS, W_ELEMS / 8);
  k_cvt8<<<512, 256, 0, stream>>>(Wv, wbf + 2 * W_ELEMS, W_ELEMS / 8);
  const size_t zeroN = (size_t)B_ * H_ * D_ * D_ + (size_t)B_ * H_ * D_;
  k_zero<<<1040, 256, 0, stream>>>(kv, zeroN);

  // K1: fused QKV projection + normalize/mask (TDM-staged x tiles)
  k_qkv<<<dim3(B_ * (N_ / 64), H_), 256, 0, stream>>>(xbf, wbf, mask,
                                                      qbf, kbf, vf);
  // K2: kv = k^T v (+ ks_sum), partial chunks with f32 atomics
  k_kv<<<dim3(B_ * H_, 64), 256, 0, stream>>>(kbf, vf, kv, ksum);
  // K3: output (TDM-staged q tiles)
  k_out<<<B_ * (N_ / 64), 256, 0, stream>>>(qbf, vf, kv, ksum, out);
}
